// TemporalAttentionBlock_8890582303466
// MI455X (gfx1250) — matrix operand order repaired and checked by reading, not compiled, not run on hardware
//
#include <hip/hip_runtime.h>
#include <stdint.h>

// ---------------- problem constants ----------------
constexpr int Bc   = 8;
constexpr int Tc   = 512;
constexpr int Vc   = 24;
constexpr int Dc   = 512;
constexpr int Hc   = 8;
constexpr int DHc  = 64;     // D / H
constexpr int DFFc = 2048;
constexpr int BVc  = Bc * Vc;          // 192
constexpr int NR   = BVc * Tc;         // 98304 rows of the (BV*T, D) activation matrix

// ---------------- WMMA types ----------------
typedef __attribute__((ext_vector_type(16))) __bf16 v16bf;
typedef __attribute__((ext_vector_type(8)))  float  v8f;

union Frag16 { v16bf v; unsigned int w[8]; unsigned short u[16]; };
union Acc8   { v8f   v; float f[8]; };

__device__ __forceinline__ v8f wmma_bf16f32(const v16bf& a, const v16bf& b, v8f c) {
  // D = A(16x32 bf16) x B(32x16 bf16) + C(16x16 f32)
  return __builtin_amdgcn_wmma_f32_16x16x32_bf16(false, a, false, b, (short)0, c, false, false);
}

// round-to-nearest-even f32 -> bf16 (as raw u16)
__device__ __forceinline__ unsigned short f2bf_u16(float x) {
  unsigned int u = __builtin_bit_cast(unsigned int, x);
  unsigned int r = u + 0x7FFFu + ((u >> 16) & 1u);
  return (unsigned short)(r >> 16);
}

__device__ __forceinline__ float gelu_exact(float x) {
  return 0.5f * x * (1.0f + erff(x * 0.70710678118654752f));
}

// 16-bit A-matrix 16x32 fragment K index for VGPR i, lane-half hi (contiguous: k = 16*hi + 2*i)
__device__ __forceinline__ int a_k_of(int i, int hi) {
  return 16 * hi + 2 * i;
}

// ---------------- f32 -> bf16 conversion ----------------
__global__ __launch_bounds__(256)
void f32_to_bf16_kernel(const float* __restrict__ in, unsigned short* __restrict__ out, int n) {
  int i = blockIdx.x * 256 + threadIdx.x;
  if (i < n) out[i] = f2bf_u16(in[i]);
}

// ---------------- LayerNorm (one block = one row of 512) ----------------
// FROM_X: read x in (B,T,V,D) layout with (BV,T) row mapping; else read row-major f32.
template<bool FROM_X>
__global__ __launch_bounds__(128)
void layernorm_kernel(const float* __restrict__ src,
                      const float* __restrict__ gamma,
                      const float* __restrict__ beta,
                      unsigned short* __restrict__ out) {
  __shared__ float s1[128];
  __shared__ float s2[128];
  const int row = blockIdx.x;
  const int tid = threadIdx.x;
  size_t base;
  if (FROM_X) {
    const int bvi = row / Tc, t = row % Tc;
    const int bb = bvi / Vc, vv = bvi % Vc;
    base = (((size_t)bb * Tc + t) * Vc + vv) * Dc;
  } else {
    base = (size_t)row * Dc;
  }
  float4 xv = *(const float4*)(src + base + tid * 4);
  s1[tid] = xv.x + xv.y + xv.z + xv.w;
  s2[tid] = xv.x * xv.x + xv.y * xv.y + xv.z * xv.z + xv.w * xv.w;
  __syncthreads();
  for (int off = 64; off > 0; off >>= 1) {
    if (tid < off) { s1[tid] += s1[tid + off]; s2[tid] += s2[tid + off]; }
    __syncthreads();
  }
  const float mean = s1[0] * (1.0f / Dc);
  const float var  = s2[0] * (1.0f / Dc) - mean * mean;
  const float rstd = rsqrtf(var + 1e-5f);
  const int c = tid * 4;
  float xs[4] = {xv.x, xv.y, xv.z, xv.w};
#pragma unroll
  for (int j = 0; j < 4; ++j) {
    float y = (xs[j] - mean) * rstd * gamma[c + j] + beta[c + j];
    out[(size_t)row * Dc + c + j] = f2bf_u16(y);
  }
}

// ---------------- WMMA GEMM: C(MxN,f32) = A(MxK,bf16) @ B(KxN,bf16) + bias ----------------
// Block tile 128x128, 8 waves, each wave a 32x64 tile (2x4 WMMA 16x16), K-step 32.
// Register-stage pipelining: global loads for slice k+1 issue before WMMAs of slice k.
// EPI: 0 = bf16 out; 1 = gelu->bf16 out; 2 = f32 out (+ residual read from x transposed);
//      3 = f32 out written to x-transposed layout (+ residual from row-major f32).
template<int EPI>
__global__ __launch_bounds__(256)
void wmma_gemm_kernel(const unsigned short* __restrict__ A,
                      const unsigned short* __restrict__ Bw,
                      const float* __restrict__ bias,
                      unsigned short* __restrict__ outb,
                      float* __restrict__ outf,
                      const float* __restrict__ resid,
                      int M, int N, int K) {
  constexpr int LSTR = 40;                       // u16 stride (80B: 16B-aligned rows, padded)
  __shared__ unsigned short As[128 * LSTR];      // 128(m) x 32(k) tile of A
  __shared__ unsigned short Bs[128 * LSTR];      // 128(n) x 32(k); B transposed so K is contiguous

  const int tid  = threadIdx.x;
  const int lane = tid & 31;
  const int wid  = tid >> 5;
  const int wm   = wid & 3;                      // 4 waves along M (32 rows each)
  const int wn   = wid >> 2;                     // 2 waves along N (64 cols each)
  const int m_blk = blockIdx.y * 128;
  const int n_blk = blockIdx.x * 128;
  const int mm = lane & 15;
  const int hi = lane >> 4;

  Acc8 acc[2][4];
#pragma unroll
  for (int a = 0; a < 2; ++a)
#pragma unroll
    for (int b = 0; b < 4; ++b)
#pragma unroll
      for (int j = 0; j < 8; ++j) acc[a][b].f[j] = 0.0f;

  const int a_row  = tid >> 1;                   // 0..127
  const int a_half = tid & 1;                    // which 16-element half of the 32-K row
  const int b_kr   = tid >> 3;                   // 0..31 (k row)
  const int b_ng   = (tid & 7) * 16;             // 0..112 (n group of 16)

  uint4 pa0, pa1, pb0, pb1;                      // register prefetch stage

  auto ld_global = [&](int k0) {
    const uint4* asrc = (const uint4*)(A + (size_t)(m_blk + a_row) * K + k0 + a_half * 16);
    pa0 = asrc[0];
    pa1 = asrc[1];
    const uint4* bsrc = (const uint4*)(Bw + (size_t)(k0 + b_kr) * N + n_blk + b_ng);
    pb0 = bsrc[0];
    pb1 = bsrc[1];
  };
  auto st_lds = [&]() {
    *(uint4*)(As + a_row * LSTR + a_half * 16)     = pa0;
    *(uint4*)(As + a_row * LSTR + a_half * 16 + 8) = pa1;
    union { uint4 q; unsigned short s[8]; } t0, t1;
    t0.q = pb0; t1.q = pb1;
#pragma unroll
    for (int j = 0; j < 8; ++j) {
      Bs[(b_ng + j) * LSTR + b_kr]     = t0.s[j];
      Bs[(b_ng + 8 + j) * LSTR + b_kr] = t1.s[j];
    }
  };
  auto compute = [&]() {
    Frag16 af[2], bfr[4];
#pragma unroll
    for (int ms = 0; ms < 2; ++ms) {
      const unsigned short* p = As + (wm * 32 + ms * 16 + mm) * LSTR;
#pragma unroll
      for (int i = 0; i < 8; ++i)
        af[ms].w[i] = *(const unsigned int*)(p + a_k_of(i, hi));
    }
#pragma unroll
    for (int ns = 0; ns < 4; ++ns) {
      const unsigned short* p = Bs + (wn * 64 + ns * 16 + mm) * LSTR;
#pragma unroll
      for (int i = 0; i < 8; ++i)
        bfr[ns].w[i] = *(const unsigned int*)(p + 16 * hi + 2 * i);
    }
#pragma unroll
    for (int ms = 0; ms < 2; ++ms)
#pragma unroll
      for (int ns = 0; ns < 4; ++ns)
        acc[ms][ns].v = wmma_bf16f32(af[ms].v, bfr[ns].v, acc[ms][ns].v);
  };

  // software pipeline over K slices of 32
  ld_global(0);
  st_lds();
  __syncthreads();
  for (int k0 = 32; k0 < K; k0 += 32) {
    ld_global(k0);       // next slice -> registers (overlaps with WMMA below)
    compute();           // consume current LDS contents
    __syncthreads();
    st_lds();
    __syncthreads();
  }
  compute();

  // epilogue: C-layout is row = j + 8*hi, col = lane&15
#pragma unroll
  for (int ms = 0; ms < 2; ++ms) {
#pragma unroll
    for (int ns = 0; ns < 4; ++ns) {
      const int col = n_blk + wn * 64 + ns * 16 + mm;
      const float bc = bias[col];
#pragma unroll
      for (int j = 0; j < 8; ++j) {
        const int row = m_blk + wm * 32 + ms * 16 + j + 8 * hi;
        float val = acc[ms][ns].f[j] + bc;
        if (EPI == 0) {
          outb[(size_t)row * N + col] = f2bf_u16(val);
        } else if (EPI == 1) {
          outb[(size_t)row * N + col] = f2bf_u16(gelu_exact(val));
        } else if (EPI == 2) {
          const int bvi = row / Tc, t = row % Tc;
          const int bb = bvi / Vc, vv = bvi % Vc;
          const size_t xi = (((size_t)bb * Tc + t) * Vc + vv) * Dc + col;
          outf[(size_t)row * N + col] = val + resid[xi];
        } else {
          const int bvi = row / Tc, t = row % Tc;
          const int bb = bvi / Vc, vv = bvi % Vc;
          const size_t xi = (((size_t)bb * Tc + t) * Vc + vv) * Dc + col;
          outf[xi] = val + resid[(size_t)row * N + col];
        }
      }
    }
  }
}

// ---------------- flash attention: 1 wave = 16 query rows of one (bv,h) ----------------
__global__ __launch_bounds__(128)
void flash_attention_kernel(const unsigned short* __restrict__ Q,
                            const unsigned short* __restrict__ Km,
                            const unsigned short* __restrict__ Vm,
                            const unsigned char* __restrict__ mask,
                            unsigned short* __restrict__ O) {
  __shared__ unsigned short lds_p[4][16 * 40];   // per-wave P transpose staging
  const int tid  = threadIdx.x;
  const int lane = tid & 31;
  const int w    = tid >> 5;
  const int mm   = lane & 15;
  const int hi   = lane >> 4;
  const float NEGINF = -__builtin_inff();

  int idx = blockIdx.x;
  const int MT = Tc / 64;
  const int mt = idx % MT; idx /= MT;
  const int h  = idx % Hc; idx /= Hc;
  const int bvi = idx;                 // 0..BV-1
  const int bb  = bvi / Vc;            // batch index for the mask
  const int q0  = mt * 64 + w * 16;
  const size_t rb = (size_t)bvi * Tc;  // first row of this sequence in (BV*T, D)

  // Q tile 16x64 as two A-fragments (K chunks of 32); feature dim is contiguous
  Frag16 qa[2];
#pragma unroll
  for (int c = 0; c < 2; ++c) {
    const unsigned short* qp = Q + (rb + q0 + mm) * Dc + h * DHc + c * 32;
#pragma unroll
    for (int i = 0; i < 8; ++i)
      qa[c].w[i] = *(const unsigned int*)(qp + a_k_of(i, hi));
  }

  Acc8 o[4];                           // O tile 16x64 (4 n-chunks of 16)
  float mrun[8], lrun[8];
#pragma unroll
  for (int nc = 0; nc < 4; ++nc)
#pragma unroll
    for (int j = 0; j < 8; ++j) o[nc].f[j] = 0.0f;
#pragma unroll
  for (int j = 0; j < 8; ++j) { mrun[j] = NEGINF; lrun[j] = 0.0f; }

  const float rscale = 0.125f;         // 1/sqrt(DH)

  for (int s0 = 0; s0 < Tc; s0 += 32) {
    // --- scores for 32 keys: two 16x16 frags, K-reduction over DH=64 (2 wmma each) ---
    Acc8 sc[2];
#pragma unroll
    for (int kt = 0; kt < 2; ++kt) {
#pragma unroll
      for (int j = 0; j < 8; ++j) sc[kt].f[j] = 0.0f;
      // B = K^T: lane n -> key s0+kt*16+n, fragment K pairs contiguous along DH
      const unsigned short* kp = Km + (rb + s0 + kt * 16 + mm) * Dc + h * DHc;
#pragma unroll
      for (int c = 0; c < 2; ++c) {
        Frag16 kb;
#pragma unroll
        for (int i = 0; i < 8; ++i)
          kb.w[i] = *(const unsigned int*)(kp + c * 32 + 16 * hi + 2 * i);
        sc[kt].v = wmma_bf16f32(qa[c].v, kb.v, sc[kt].v);
      }
      // scale + key mask (column-wise: this lane owns one key column)
      const int skey = s0 + kt * 16 + mm;
      const bool valid = mask[bb * Tc + skey] != 0;
#pragma unroll
      for (int j = 0; j < 8; ++j)
        sc[kt].f[j] = valid ? sc[kt].f[j] * rscale : NEGINF;
    }

    // --- online softmax: row r = j + 8*hi lives in 16 lanes of one half ---
#pragma unroll
    for (int j = 0; j < 8; ++j) {
      float vmax = fmaxf(sc[0].f[j], sc[1].f[j]);
#pragma unroll
      for (int off2 = 1; off2 < 16; off2 <<= 1)
        vmax = fmaxf(vmax, __shfl_xor(vmax, off2, 16));
      const float mnew = fmaxf(mrun[j], vmax);
      const float corr = (mnew == NEGINF) ? 0.0f : __expf(mrun[j] - mnew);
      float rs = 0.0f;
#pragma unroll
      for (int kt = 0; kt < 2; ++kt) {
        const float sv = sc[kt].f[j];
        const float pv = (sv == NEGINF) ? 0.0f : __expf(sv - mnew);
        sc[kt].f[j] = pv;
        rs += pv;
      }
#pragma unroll
      for (int off2 = 1; off2 < 16; off2 <<= 1)
        rs += __shfl_xor(rs, off2, 16);
      lrun[j] = lrun[j] * corr + rs;
      mrun[j] = mnew;
#pragma unroll
      for (int nc = 0; nc < 4; ++nc) o[nc].f[j] *= corr;
    }

    // --- P: C-layout -> LDS row-major 16x32 -> A-layout fragment (bf16) ---
    unsigned short* lp = lds_p[w];
#pragma unroll
    for (int kt = 0; kt < 2; ++kt)
#pragma unroll
      for (int j = 0; j < 8; ++j)
        lp[(j + 8 * hi) * 40 + kt * 16 + mm] = f2bf_u16(sc[kt].f[j]);
    asm volatile("s_wait_dscnt 0" ::: "memory");   // intra-wave LDS RAW
    Frag16 pa;
#pragma unroll
    for (int i = 0; i < 8; ++i)
      pa.w[i] = *(const unsigned int*)(lp + mm * 40 + a_k_of(i, hi));

    // --- O += P(16x32) @ V(32x64): 4 wmma over DH chunks ---
#pragma unroll
    for (int nc = 0; nc < 4; ++nc) {
      Frag16 vb;
      // B: lane n -> dh = nc*16+n; fragment k = key index (stride D in memory)
      const unsigned short* vp = Vm + (rb + s0 + 16 * hi) * Dc + h * DHc + nc * 16 + mm;
#pragma unroll
      for (int i = 0; i < 8; ++i) {
        vb.u[2 * i]     = vp[(size_t)(2 * i) * Dc];
        vb.u[2 * i + 1] = vp[(size_t)(2 * i + 1) * Dc];
      }
      o[nc].v = wmma_bf16f32(pa.v, vb.v, o[nc].v);
    }
  }

  // normalize and store bf16 to (BV*T, D)
  float inv[8];
#pragma unroll
  for (int j = 0; j < 8; ++j) inv[j] = (lrun[j] > 0.0f) ? 1.0f / lrun[j] : 0.0f;
#pragma unroll
  for (int nc = 0; nc < 4; ++nc)
#pragma unroll
    for (int j = 0; j < 8; ++j) {
      const size_t orow = rb + q0 + j + 8 * hi;
      O[orow * Dc + h * DHc + nc * 16 + mm] = f2bf_u16(o[nc].f[j] * inv[j]);
    }
}

// ---------------- host-side orchestration ----------------
extern "C" void kernel_launch(void* const* d_in, const int* in_sizes, int n_in,
                              void* d_out, int out_size, void* d_ws, size_t ws_size,
                              hipStream_t stream) {
  (void)in_sizes; (void)n_in; (void)out_size; (void)ws_size;
  const float* x    = (const float*)d_in[0];
  const unsigned char* mask = (const unsigned char*)d_in[1];   // bool (B,T)
  const float* Wq = (const float*)d_in[2];  const float* bq = (const float*)d_in[3];
  const float* Wk = (const float*)d_in[4];  const float* bk = (const float*)d_in[5];
  const float* Wv = (const float*)d_in[6];  const float* bvp = (const float*)d_in[7];
  const float* Wo = (const float*)d_in[8];  const float* bo = (const float*)d_in[9];
  const float* W1 = (const float*)d_in[10]; const float* b1 = (const float*)d_in[11];
  const float* W2 = (const float*)d_in[12]; const float* b2 = (const float*)d_in[13];
  const float* g1 = (const float*)d_in[14]; const float* be1 = (const float*)d_in[15];
  const float* g2 = (const float*)d_in[16]; const float* be2 = (const float*)d_in[17];
  float* out = (float*)d_out;

  size_t off = 0;
  auto alloc = [&](size_t bytes) -> void* {
    void* p = (char*)d_ws + off;
    off += (bytes + 255) & ~(size_t)255;
    return p;
  };
  unsigned short* xn  = (unsigned short*)alloc((size_t)NR * Dc * 2);   // LN output (reused for LN2)
  unsigned short* Qb  = (unsigned short*)alloc((size_t)NR * Dc * 2);
  unsigned short* Kb  = (unsigned short*)alloc((size_t)NR * Dc * 2);
  unsigned short* Vb  = (unsigned short*)alloc((size_t)NR * Dc * 2);
  unsigned short* Ab  = (unsigned short*)alloc((size_t)NR * Dc * 2);   // attention output
  float*          xr  = (float*)alloc((size_t)NR * Dc * 4);            // residual stream (f32)
  unsigned short* ffb = (unsigned short*)alloc((size_t)NR * DFFc * 2); // gelu(h@W1+b1)
  unsigned short* wqb = (unsigned short*)alloc((size_t)Dc * Dc * 2);
  unsigned short* wkb = (unsigned short*)alloc((size_t)Dc * Dc * 2);
  unsigned short* wvb = (unsigned short*)alloc((size_t)Dc * Dc * 2);
  unsigned short* wob = (unsigned short*)alloc((size_t)Dc * Dc * 2);
  unsigned short* w1b = (unsigned short*)alloc((size_t)Dc * DFFc * 2);
  unsigned short* w2b = (unsigned short*)alloc((size_t)DFFc * Dc * 2);

  // weights -> bf16
  const int nDD = Dc * Dc, nDF = Dc * DFFc;
  f32_to_bf16_kernel<<<(nDD + 255) / 256, 256, 0, stream>>>(Wq, wqb, nDD);
  f32_to_bf16_kernel<<<(nDD + 255) / 256, 256, 0, stream>>>(Wk, wkb, nDD);
  f32_to_bf16_kernel<<<(nDD + 255) / 256, 256, 0, stream>>>(Wv, wvb, nDD);
  f32_to_bf16_kernel<<<(nDD + 255) / 256, 256, 0, stream>>>(Wo, wob, nDD);
  f32_to_bf16_kernel<<<(nDF + 255) / 256, 256, 0, stream>>>(W1, w1b, nDF);
  f32_to_bf16_kernel<<<(nDF + 255) / 256, 256, 0, stream>>>(W2, w2b, nDF);

  // LN1 (reads x transposed)
  layernorm_kernel<true><<<NR, 128, 0, stream>>>(x, g1, be1, xn);

  // Q, K, V projections
  dim3 gD(Dc / 128, NR / 128);
  wmma_gemm_kernel<0><<<gD, 256, 0, stream>>>(xn, wqb, bq, Qb, nullptr, nullptr, NR, Dc, Dc);
  wmma_gemm_kernel<0><<<gD, 256, 0, stream>>>(xn, wkb, bk, Kb, nullptr, nullptr, NR, Dc, Dc);
  wmma_gemm_kernel<0><<<gD, 256, 0, stream>>>(xn, wvb, bvp, Vb, nullptr, nullptr, NR, Dc, Dc);

  // attention
  flash_attention_kernel<<<BVc * Hc * (Tc / 64), 128, 0, stream>>>(Qb, Kb, Vb, mask, Ab);

  // output projection + residual (xr = x + attn@Wo + bo), f32
  wmma_gemm_kernel<2><<<gD, 256, 0, stream>>>(Ab, wob, bo, nullptr, xr, x, NR, Dc, Dc);

  // LN2
  layernorm_kernel<false><<<NR, 128, 0, stream>>>(xr, g2, be2, xn);

  // FFN up + exact GELU
  dim3 gF(DFFc / 128, NR / 128);
  wmma_gemm_kernel<1><<<gF, 256, 0, stream>>>(xn, w1b, b1, ffb, nullptr, nullptr, NR, DFFc, Dc);

  // FFN down + residual, written transposed back to (B,T,V,D)
  wmma_gemm_kernel<3><<<gD, 256, 0, stream>>>(ffb, w2b, b2, nullptr, out, xr, NR, Dc, DFFc);
}
